// UniGCNIILayer_910533067196
// MI455X (gfx1250) — matrix operand order repaired and checked by reading, not compiled, not run on hardware
//
#include <hip/hip_runtime.h>
#include <hip/hip_bf16.h>

// UniGCNII layer for MI455X (gfx1250): bf16 WMMA GEMMs with hi/lo split for
// f32-accuracy, degree reductions fused into GEMM1, wave32 / 8-wave blocks,
// Tensor Data Mover (TDM) double-buffered staging for the incidence stream.

#define NN 8192   // nodes
#define NE 8192   // edges
#define CD 256    // channels
#define ALPHA 0.1f
#define BETA  0.5f
#define P 40      // LDS row pitch (elements): 80B rows, fragment segments 16B-aligned

typedef __bf16 bf16_t;
typedef __attribute__((ext_vector_type(16))) __bf16 v16bf;
typedef __attribute__((ext_vector_type(8)))  __bf16 v8bf;
typedef __attribute__((ext_vector_type(8)))  float  v8f;
typedef __attribute__((ext_vector_type(4)))  unsigned int v4u;
typedef __attribute__((ext_vector_type(8)))  int v8i;
typedef __attribute__((ext_vector_type(4)))  int v4i;

#if defined(__has_builtin)
#if __has_builtin(__builtin_amdgcn_tensor_load_to_lds) && \
    __has_builtin(__builtin_amdgcn_s_wait_tensorcnt)
#define HAVE_TDM 1
#endif
#endif
#ifndef HAVE_TDM
#define HAVE_TDM 0
#endif

__device__ __forceinline__ v16bf make_frag(const bf16_t* p0, const bf16_t* p1) {
  v8bf lo = *(const v8bf*)p0;
  v8bf hi = *(const v8bf*)p1;
  v16bf f;
#pragma unroll
  for (int i = 0; i < 8; ++i) { f[i] = lo[i]; f[i + 8] = hi[i]; }
  return f;
}

// A fragment (16x32 bf16, row-major in LDS): lane row = lane%16, half hf
// holds k = hf*16 + v for elems v = 0..15 (16 contiguous bf16).
__device__ __forceinline__ v16bf load_a_frag(const bf16_t* a_s, int row, int hf) {
  const bf16_t* base = a_s + row * P;
  return make_frag(base + hf * 16, base + hf * 16 + 8);
}
// B fragment (32x16 bf16): stored k-transposed b_s[col][k]; lane col = lane%16,
// k = hf*16 + v -> 16 contiguous bf16.
__device__ __forceinline__ v16bf load_b_frag(const bf16_t* b_s, int col, int hf) {
  const bf16_t* base = b_s + col * P + hf * 16;
  return make_frag(base, base + 8);
}

__device__ __forceinline__ void split2(float v, bf16_t& h, bf16_t& l) {
  h = (bf16_t)v;
  l = (bf16_t)(v - (float)h);
}

#define WMMA_BF16(accv, a, b)                                                  \
  accv = __builtin_amdgcn_wmma_f32_16x16x32_bf16(false, (a), false, (b),       \
                                                 (short)0, (accv), false, false)

#if HAVE_TDM
// Issue a TDM 2-D tile load: tile_h rows x tile_w f32 (row stride `stride`
// elements) from global `g` into LDS byte address `lds_addr` (contiguous).
// D# bit layout per CDNA5 ISA ch.8 (group0/group1; groups 2,3 zero for 2-D).
// This toolchain exposes the 6-arg builtin: (v4u, v8i, v4i, v4i, v8i, cpol).
__device__ __forceinline__ void tdm_load_2d_f32(const float* g, unsigned lds_addr,
                                                int tile_w, int tile_h, int stride) {
  const unsigned long long ga = (unsigned long long)(uintptr_t)g;
  v4u g0;
  g0.x = 1u;                                                 // count=1, user D#
  g0.y = lds_addr;                                           // lds_addr (bytes)
  g0.z = (unsigned)ga;                                       // global_addr[31:0]
  g0.w = (unsigned)((ga >> 32) & 0x01ffffffu) | (2u << 30);  // addr[56:32] | type=2
  const unsigned td0 = (unsigned)stride;                     // tensor_dim0
  const unsigned td1 = 1u << 20;                             // tensor_dim1 (ample)
  v8i g1;
  g1[0] = (int)(2u << 16);                                   // data_size=4B, mask=0
  g1[1] = (int)(td0 << 16);                                  // tensor_dim0[15:0]
  g1[2] = (int)((td0 >> 16) | (td1 << 16));                  // td0[31:16]|td1[15:0]
  g1[3] = (int)((td1 >> 16) | ((unsigned)tile_w << 16));     // td1[31:16]|tile_dim0
  g1[4] = tile_h;                                            // tile_dim1
  g1[5] = stride;                                            // dim0_stride[31:0]
  g1[6] = 0;
  g1[7] = 0;
  const v4i z4 = {0, 0, 0, 0};
  const v8i z8 = {0, 0, 0, 0, 0, 0, 0, 0};
  __builtin_amdgcn_tensor_load_to_lds(g0, g1, z4, z4, z8, 0);
}
#endif

// ---------------------------------------------------------------- node degree
__global__ void __launch_bounds__(256)
k_rowsum(const float* __restrict__ inc, float* __restrict__ nd,
         float* __restrict__ isnd) {
  __shared__ float red[256];
  const int n = blockIdx.x;
  float s = 0.f;
  for (int e = threadIdx.x; e < NE; e += 256) s += inc[(size_t)n * NE + e];
  red[threadIdx.x] = s;
  __syncthreads();
  for (int off = 128; off > 0; off >>= 1) {
    if (threadIdx.x < off) red[threadIdx.x] += red[threadIdx.x + off];
    __syncthreads();
  }
  if (threadIdx.x == 0) {
    nd[n] = red[0];
    isnd[n] = rsqrtf(red[0]);
  }
}

// ------------------------- GEMM1: m01 = B^T x, fused colsum/wsum + edge scale
__global__ void __launch_bounds__(256)
k_gemm1(const float* __restrict__ inc, const float* __restrict__ x,
        const float* __restrict__ nd,
        bf16_t* __restrict__ m01h, bf16_t* __restrict__ m01l) {
  __shared__ alignas(16) float  tdm_s[2 * 32 * 64];  // staging: [buf][n(32)][e(64)]
  __shared__ alignas(16) bf16_t a_s[64 * P];
  __shared__ alignas(16) bf16_t bh_s[256 * P];
  __shared__ alignas(16) bf16_t bl_s[256 * P];
  __shared__ float red_s[2][256];
  __shared__ float es_s[64];

  const int t = threadIdx.x;
  const int e0 = blockIdx.x * 64;
  const int lane = t & 31, wid = t >> 5;
  const int wm = wid & 3, wn = wid >> 2;
  const int hf = lane >> 4, l16 = lane & 15;
  const int eload = t & 63, nbase = t >> 6;

  v8f acc[8];
#pragma unroll
  for (int i = 0; i < 8; ++i)
#pragma unroll
    for (int j = 0; j < 8; ++j) acc[i][j] = 0.f;

  float csum = 0.f, wsum = 0.f;

#if HAVE_TDM
  const unsigned lds0 = (unsigned)(uintptr_t)&tdm_s[0];
  const unsigned lds1 = (unsigned)(uintptr_t)&tdm_s[32 * 64];
  if (wid == 0) tdm_load_2d_f32(inc + e0, lds0, 64, 32, NE);
#endif

  for (int kc = 0; kc < NN; kc += 32) {
#if HAVE_TDM
    if (wid == 0) __builtin_amdgcn_s_wait_tensorcnt(0);
#endif
    __syncthreads();
#if HAVE_TDM
    const float* tb = &tdm_s[((kc >> 5) & 1) * (32 * 64)];
#endif
    // A tile: B^T[e, n] = inc[n, e]; also accumulate column sums of inc.
#pragma unroll
    for (int i = 0; i < 8; ++i) {
      const int n = nbase + 4 * i;
#if HAVE_TDM
      const float v = tb[n * 64 + eload];
#else
      const float v = inc[(size_t)(kc + n) * NE + e0 + eload];
#endif
      a_s[eload * P + n] = (bf16_t)v;  // exact: entries are 0/1
      csum += v;
      wsum += v * nd[kc + n];
    }
    // B tile: x split hi/lo, stored k-transposed.
#pragma unroll
    for (int n = 0; n < 32; ++n) {
      const float v = x[(size_t)(kc + n) * CD + t];
      bf16_t h, l; split2(v, h, l);
      bh_s[t * P + n] = h;
      bl_s[t * P + n] = l;
    }
    __syncthreads();
#if HAVE_TDM
    if (wid == 0 && kc + 32 < NN)   // prefetch next A tile into the other buffer
      tdm_load_2d_f32(inc + (size_t)(kc + 32) * NE + e0,
                      ((kc >> 5) & 1) ? lds0 : lds1, 64, 32, NE);
#endif
    const v16bf af = load_a_frag(a_s, wm * 16 + l16, hf);
#pragma unroll
    for (int tn = 0; tn < 8; ++tn) {
      const int col = wn * 128 + tn * 16 + l16;
      const v16bf bh = load_b_frag(bh_s, col, hf);
      const v16bf bl = load_b_frag(bl_s, col, hf);
      WMMA_BF16(acc[tn], af, bh);
      WMMA_BF16(acc[tn], af, bl);
    }
  }
  __syncthreads();
  red_s[0][t] = csum;
  red_s[1][t] = wsum;
  __syncthreads();
  if (t < 64) {  // this block owns its 64 edge-columns completely
    const float cs = red_s[0][t] + red_s[0][t + 64] + red_s[0][t + 128] + red_s[0][t + 192];
    const float ws = red_s[1][t] + red_s[1][t + 64] + red_s[1][t + 128] + red_s[1][t + 192];
    es_s[t] = sqrtf(cs / ws);  // 1/sqrt(edge_degree), wsum>=1 (self-loops)
  }
  __syncthreads();
#pragma unroll
  for (int tn = 0; tn < 8; ++tn) {
    const int col = wn * 128 + tn * 16 + l16;
#pragma unroll
    for (int r = 0; r < 8; ++r) {
      const int erow = wm * 16 + hf * 8 + r;
      const float v = acc[tn][r] * es_s[erow];
      bf16_t h, l; split2(v, h, l);
      const size_t idx = (size_t)(e0 + erow) * CD + col;
      m01h[idx] = h;
      m01l[idx] = l;
    }
  }
}

// ------------------- GEMM2: m10 = B * m01_scaled; epilogue: 1/sqrt(d_n) + skip
__global__ void __launch_bounds__(256)
k_gemm2(const float* __restrict__ inc, const float* __restrict__ x,
        const float* __restrict__ isnd,
        const bf16_t* __restrict__ m01h, const bf16_t* __restrict__ m01l,
        bf16_t* __restrict__ xch, bf16_t* __restrict__ xcl) {
  __shared__ alignas(16) float  tdm_s[2 * 64 * 32];  // staging: [buf][n(64)][e(32)]
  __shared__ alignas(16) bf16_t a_s[64 * P];
  __shared__ alignas(16) bf16_t bh_s[256 * P];
  __shared__ alignas(16) bf16_t bl_s[256 * P];

  const int t = threadIdx.x;
  const int n0 = blockIdx.x * 64;
  const int lane = t & 31, wid = t >> 5;
  const int wm = wid & 3, wn = wid >> 2;
  const int hf = lane >> 4, l16 = lane & 15;
  const int kload = t & 31, nload = t >> 5;

  v8f acc[8];
#pragma unroll
  for (int i = 0; i < 8; ++i)
#pragma unroll
    for (int j = 0; j < 8; ++j) acc[i][j] = 0.f;

#if HAVE_TDM
  const unsigned lds0 = (unsigned)(uintptr_t)&tdm_s[0];
  const unsigned lds1 = (unsigned)(uintptr_t)&tdm_s[64 * 32];
  if (wid == 0) tdm_load_2d_f32(inc + (size_t)n0 * NE, lds0, 32, 64, NE);
#endif

  for (int kc = 0; kc < NE; kc += 32) {
#if HAVE_TDM
    if (wid == 0) __builtin_amdgcn_s_wait_tensorcnt(0);
#endif
    __syncthreads();
#if HAVE_TDM
    const float* tb = &tdm_s[((kc >> 5) & 1) * (64 * 32)];
#endif
#pragma unroll
    for (int i = 0; i < 8; ++i) {
      const int n = nload + 8 * i;
#if HAVE_TDM
      a_s[n * P + kload] = (bf16_t)tb[n * 32 + kload];
#else
      a_s[n * P + kload] = (bf16_t)inc[(size_t)(n0 + n) * NE + kc + kload];
#endif
    }
#pragma unroll
    for (int e = 0; e < 32; ++e) {
      const size_t idx = (size_t)(kc + e) * CD + t;
      bh_s[t * P + e] = m01h[idx];
      bl_s[t * P + e] = m01l[idx];
    }
    __syncthreads();
#if HAVE_TDM
    if (wid == 0 && kc + 32 < NE)
      tdm_load_2d_f32(inc + (size_t)n0 * NE + kc + 32,
                      ((kc >> 5) & 1) ? lds0 : lds1, 32, 64, NE);
#endif
    const v16bf af = load_a_frag(a_s, wm * 16 + l16, hf);
#pragma unroll
    for (int tn = 0; tn < 8; ++tn) {
      const int col = wn * 128 + tn * 16 + l16;
      const v16bf bh = load_b_frag(bh_s, col, hf);
      const v16bf bl = load_b_frag(bl_s, col, hf);
      WMMA_BF16(acc[tn], af, bh);
      WMMA_BF16(acc[tn], af, bl);
    }
  }
#pragma unroll
  for (int tn = 0; tn < 8; ++tn) {
    const int col = wn * 128 + tn * 16 + l16;
#pragma unroll
    for (int r = 0; r < 8; ++r) {
      const int row = n0 + wm * 16 + hf * 8 + r;
      const size_t idx = (size_t)row * CD + col;
      const float v =
          (1.0f - ALPHA) * acc[tn][r] * isnd[row] + ALPHA * x[idx];
      bf16_t h, l; split2(v, h, l);
      xch[idx] = h;
      xcl[idx] = l;
    }
  }
}

// --------------------------- GEMM3: out = (1-beta)*xc + beta*(xc @ W^T)
__global__ void __launch_bounds__(256)
k_gemm3(const bf16_t* __restrict__ xch, const bf16_t* __restrict__ xcl,
        const float* __restrict__ W, float* __restrict__ out) {
  __shared__ alignas(16) bf16_t ah_s[64 * P];
  __shared__ alignas(16) bf16_t al_s[64 * P];
  __shared__ alignas(16) bf16_t bh_s[256 * P];
  __shared__ alignas(16) bf16_t bl_s[256 * P];

  const int t = threadIdx.x;
  const int n0 = blockIdx.x * 64;
  const int lane = t & 31, wid = t >> 5;
  const int wm = wid & 3, wn = wid >> 2;
  const int hf = lane >> 4, l16 = lane & 15;
  const int kload = t & 31, nload = t >> 5;

  v8f acc[8];
#pragma unroll
  for (int i = 0; i < 8; ++i)
#pragma unroll
    for (int j = 0; j < 8; ++j) acc[i][j] = 0.f;

  for (int kc = 0; kc < CD; kc += 32) {
    __syncthreads();
#pragma unroll
    for (int i = 0; i < 8; ++i) {
      const int n = nload + 8 * i;
      const size_t idx = (size_t)(n0 + n) * CD + kc + kload;
      ah_s[n * P + kload] = xch[idx];
      al_s[n * P + kload] = xcl[idx];
    }
#pragma unroll
    for (int k = 0; k < 32; ++k) {
      const float v = W[(size_t)t * CD + kc + k];  // B[k][c] = W[c][k]
      bf16_t h, l; split2(v, h, l);
      bh_s[t * P + k] = h;
      bl_s[t * P + k] = l;
    }
    __syncthreads();
    const v16bf afh = load_a_frag(ah_s, wm * 16 + l16, hf);
    const v16bf afl = load_a_frag(al_s, wm * 16 + l16, hf);
#pragma unroll
    for (int tn = 0; tn < 8; ++tn) {
      const int col = wn * 128 + tn * 16 + l16;
      const v16bf bh = load_b_frag(bh_s, col, hf);
      const v16bf bl = load_b_frag(bl_s, col, hf);
      WMMA_BF16(acc[tn], afh, bh);
      WMMA_BF16(acc[tn], afh, bl);  // hi x lo
      WMMA_BF16(acc[tn], afl, bh);  // lo x hi (lo x lo negligible)
    }
  }
#pragma unroll
  for (int tn = 0; tn < 8; ++tn) {
    const int col = wn * 128 + tn * 16 + l16;
#pragma unroll
    for (int r = 0; r < 8; ++r) {
      const int row = n0 + wm * 16 + hf * 8 + r;
      const size_t idx = (size_t)row * CD + col;
      const float xc = (float)xch[idx] + (float)xcl[idx];  // ~f32-accurate
      out[idx] = (1.0f - BETA) * xc + BETA * acc[tn][r];
    }
  }
}

// -------------------------------------------------------------------- launch
extern "C" void kernel_launch(void* const* d_in, const int* in_sizes, int n_in,
                              void* d_out, int out_size, void* d_ws,
                              size_t ws_size, hipStream_t stream) {
  (void)in_sizes; (void)n_in; (void)out_size; (void)ws_size;
  const float* x   = (const float*)d_in[0];  // [N, C]
  const float* inc = (const float*)d_in[1];  // [N, E]
  const float* W   = (const float*)d_in[2];  // [C, C]
  float* out = (float*)d_out;                // [N, C]

  // workspace layout (~16.1 MB)
  float*  nd   = (float*)d_ws;                     // [N] node degree
  float*  isnd = nd + NN;                          // [N] 1/sqrt(degree)
  bf16_t* m01h = (bf16_t*)(isnd + NN);             // [E, C] scaled m01 hi
  bf16_t* m01l = m01h + (size_t)NE * CD;           // [E, C] scaled m01 lo
  bf16_t* xch  = m01l + (size_t)NE * CD;           // [N, C] x_combined hi
  bf16_t* xcl  = xch + (size_t)NN * CD;            // [N, C] x_combined lo

  k_rowsum<<<NN, 256, 0, stream>>>(inc, nd, isnd);
  k_gemm1<<<NE / 64, 256, 0, stream>>>(inc, x, nd, m01h, m01l);
  k_gemm2<<<NN / 64, 256, 0, stream>>>(inc, x, isnd, m01h, m01l, xch, xcl);
  k_gemm3<<<NN / 64, 256, 0, stream>>>(xch, xcl, W, out);
}
